// my_CE_logexp_26336739459493
// MI455X (gfx1250) — matrix operand order repaired and checked by reading, not compiled, not run on hardware
//
#include <hip/hip_runtime.h>
#include <hip/hip_bf16.h>
#include <math.h>

// ---------------------------------------------------------------------------
// ArcFace-style margin loss, fused for MI455X (gfx1250, wave32, WMMA + TDM).
// B=512 batch, C=100000 classes, D=512 dim.
// ---------------------------------------------------------------------------

#define B_SZ 512
#define C_SZ 100000
#define D_SZ 512

#define COS_M 0.8775825618903728f   // cos(0.5)
#define SIN_M 0.4794255386042030f   // sin(0.5)

typedef _Float16 v16h __attribute__((ext_vector_type(16)));
typedef _Float16 h8   __attribute__((ext_vector_type(8)));
typedef float    v8f  __attribute__((ext_vector_type(8)));
typedef float    f4   __attribute__((ext_vector_type(4)));
typedef unsigned int u32x4 __attribute__((ext_vector_type(4)));
typedef unsigned int u32x8 __attribute__((ext_vector_type(8)));

__device__ __forceinline__ float wave_reduce_sum(float v) {
#pragma unroll
  for (int m = 16; m > 0; m >>= 1) v += __shfl_xor(v, m, 32);
  return v;
}

// ---------------------------------------------------------------------------
// TDM: DMA one 2D weight tile (rem_rows-limited 64 rows x 32 halfs) into LDS.
// D# per CDNA5 ISA ch.8: 2-group form (tensor <= 2D, VADDR2/3 = NULL).
// OOB rows (beyond tensor_dim1) are zero-filled by hardware.
// ---------------------------------------------------------------------------
__device__ __forceinline__ void tdm_load_w_tile(unsigned int lds_addr,
                                                unsigned long long gaddr,
                                                unsigned int rem_rows) {
  u32x4 g0;
  g0[0] = 1u;                                    // count=1 (valid descriptor)
  g0[1] = lds_addr;                              // LDS byte address of tile
  g0[2] = (unsigned int)gaddr;                   // global_addr[31:0]
  g0[3] = ((unsigned int)(gaddr >> 32) & 0x01FFFFFFu) | (2u << 30);  // addr hi | type=2
  u32x8 g1;
  g1[0] = 1u << 16;                              // workgroup_mask=0, data_size=1 (2B)
  g1[1] = ((unsigned int)D_SZ & 0xFFFFu) << 16;  // tensor_dim0 lo16 (=512)
  g1[2] = ((unsigned int)D_SZ >> 16) | ((rem_rows & 0xFFFFu) << 16);  // dim0 hi | dim1 lo
  g1[3] = (rem_rows >> 16) | (32u << 16);        // dim1 hi | tile_dim0=32 elems
  g1[4] = 64u;                                   // tile_dim1=64 rows, tile_dim2=0
  g1[5] = (unsigned int)D_SZ;                    // tensor_dim0_stride lo32 (=512 elems)
  g1[6] = 0u;                                    // stride0 hi16 | stride1 lo16
  g1[7] = 0u;                                    // stride1 hi
  asm volatile("tensor_load_to_lds %0, %1" :: "s"(g0), "s"(g1) : "memory");
}

// ---------------------------------------------------------------------------
// Kernel 1: normalize embeddings -> f32 copy (for w_m math) + f16 copy (GEMM A)
// ---------------------------------------------------------------------------
__global__ void prep_e_kernel(const float* __restrict__ emb,
                              float* __restrict__ eN,
                              _Float16* __restrict__ eH) {
  const int wv = threadIdx.x >> 5, lane = threadIdx.x & 31;
  const int row = blockIdx.x * 8 + wv;
  const float* src = emb + (size_t)row * D_SZ + lane * 16;

  float x[16];
#pragma unroll
  for (int c = 0; c < 4; ++c) {
    f4 a = *(const f4*)(src + c * 4);
#pragma unroll
    for (int i = 0; i < 4; ++i) x[c * 4 + i] = a[i];
  }
  float ss = 0.f;
#pragma unroll
  for (int i = 0; i < 16; ++i) ss += x[i] * x[i];
  ss = wave_reduce_sum(ss);
  const float inv = 1.0f / fmaxf(sqrtf(ss), 1e-12f);

  float* dN = eN + (size_t)row * D_SZ + lane * 16;
#pragma unroll
  for (int c = 0; c < 4; ++c) {
    f4 o;
#pragma unroll
    for (int i = 0; i < 4; ++i) o[i] = x[c * 4 + i] * inv;
    *(f4*)(dN + c * 4) = o;
  }
  _Float16* dH = eH + (size_t)row * D_SZ + lane * 16;
#pragma unroll
  for (int c = 0; c < 2; ++c) {
    h8 h;
#pragma unroll
    for (int i = 0; i < 8; ++i) h[i] = (_Float16)(x[c * 8 + i] * inv);
    *(h8*)(dH + c * 8) = h;
  }
}

// ---------------------------------------------------------------------------
// Kernel 2: normalize weight rows -> f16. grid 12500 x 256 (wave per row).
// ---------------------------------------------------------------------------
__global__ void prep_w_kernel(const float* __restrict__ wgt,
                              _Float16* __restrict__ wH) {
  const int wv = threadIdx.x >> 5, lane = threadIdx.x & 31;
  const int row = blockIdx.x * 8 + wv;  // 12500*8 == 100000 exactly
  const float* src = wgt + (size_t)row * D_SZ + lane * 16;

  float x[16];
#pragma unroll
  for (int c = 0; c < 4; ++c) {
    f4 a = *(const f4*)(src + c * 4);
#pragma unroll
    for (int i = 0; i < 4; ++i) x[c * 4 + i] = a[i];
  }
  float ss = 0.f;
#pragma unroll
  for (int i = 0; i < 16; ++i) ss += x[i] * x[i];
  ss = wave_reduce_sum(ss);
  const float inv = 1.0f / fmaxf(sqrtf(ss), 1e-12f);

  _Float16* dH = wH + (size_t)row * D_SZ + lane * 16;
#pragma unroll
  for (int c = 0; c < 2; ++c) {
    h8 h;
#pragma unroll
    for (int i = 0; i < 8; ++i) h[i] = (_Float16)(x[c * 8 + i] * inv);
    *(h8*)(dH + c * 8) = h;
  }
}

// ---------------------------------------------------------------------------
// Kernel 3: per-row cos_t, cos(theta+m), and w_m rows. grid 64 x 256.
// ---------------------------------------------------------------------------
__global__ void prep_wm_kernel(const float* __restrict__ eN,
                               const int* __restrict__ labels,
                               const _Float16* __restrict__ wH,
                               _Float16* __restrict__ wmH,
                               float* __restrict__ cosTM) {
  const int wv = threadIdx.x >> 5, lane = threadIdx.x & 31;
  const int b = blockIdx.x * 8 + wv;
  const int lab = labels[b];

  const float* ep = eN + (size_t)b * D_SZ + lane * 16;
  const _Float16* wp = wH + (size_t)lab * D_SZ + lane * 16;

  float e[16], wy[16];
#pragma unroll
  for (int c = 0; c < 4; ++c) {
    f4 a = *(const f4*)(ep + c * 4);
#pragma unroll
    for (int i = 0; i < 4; ++i) e[c * 4 + i] = a[i];
  }
#pragma unroll
  for (int c = 0; c < 2; ++c) {
    h8 h = *(const h8*)(wp + c * 8);
#pragma unroll
    for (int i = 0; i < 8; ++i) wy[c * 8 + i] = (float)h[i];
  }
  float d = 0.f;
#pragma unroll
  for (int i = 0; i < 16; ++i) d += e[i] * wy[i];
  d = wave_reduce_sum(d);

  const float cost  = fminf(fmaxf(d, -1.f), 1.f);
  const float sint  = sqrtf(fmaxf(1.f - cost * cost, 0.f));
  const float costm = cost * COS_M - sint * SIN_M;
  const float sintm = sint * COS_M + cost * SIN_M;
  const float invs  = 1.0f / fmaxf(sint, 1e-12f);

  _Float16* dH = wmH + (size_t)b * D_SZ + lane * 16;
#pragma unroll
  for (int c = 0; c < 2; ++c) {
    h8 h;
#pragma unroll
    for (int i = 0; i < 8; ++i) {
      const float v = (-SIN_M * e[c * 8 + i] + sintm * wy[c * 8 + i]) * invs;
      h[i] = (_Float16)v;
    }
    *(h8*)(dH + c * 8) = h;
  }
  if (lane == 0) cosTM[b] = costm;
}

// ---------------------------------------------------------------------------
// Kernel 4: zero the per-row accumulators (harness does not re-poison ws).
// ---------------------------------------------------------------------------
__global__ void zero_acc_kernel(float* __restrict__ acc) {
  acc[blockIdx.x * 256 + threadIdx.x] = 0.f;
}

// ---------------------------------------------------------------------------
// Kernel 5: fused dual GEMM + margin-softmax epilogue, double-buffered LDS.
// Block tile: 256 A-rows (128 e rows + 128 matching w_m rows) x 64 classes.
// W tiles stream in via the Tensor Data Mover (wave 0 issues, TENSORcnt
// waits, workgroup barrier publishes); A tiles via global_load + ds_store
// into the alternate buffer, overlapping the 8 WMMAs of the current step.
// ---------------------------------------------------------------------------
__global__ __launch_bounds__(256) void arc_gemm_kernel(
    const _Float16* __restrict__ eH, const _Float16* __restrict__ wmH,
    const _Float16* __restrict__ wH, const int* __restrict__ labels,
    const float* __restrict__ cosTM, float* __restrict__ acc) {
  __shared__ __attribute__((aligned(64))) unsigned char smem[256 * 64 * 4];  // 64 KB
  _Float16* Abase = (_Float16*)smem;            // 2 x (256 rows x 32 halfs) = 2 x 16 KB
  _Float16* Wbase = (_Float16*)(smem + 32768);  // 2 x (64 rows x 32 halfs)  = 2 x 4 KB
  float*    Fsh   = (float*)smem;               // epilogue: 256 x 64 f32 = 64 KB

  const int tid  = threadIdx.x;
  const int cb   = blockIdx.x;       // class tile, 0..1562
  const int rb   = blockIdx.y;       // batch tile, 0..3
  const int wv   = tid >> 5, lane = tid & 31;
  const int lr   = lane & 15, lh = lane >> 4;
  const int part = wv >> 2;          // 0: e->logits, 1: w_m->sim
  const int m0   = (wv & 3) * 32;

  const _Float16* asrc = (tid < 128)
      ? (eH  + (size_t)(rb * 128 + tid)       * D_SZ)
      : (wmH + (size_t)(rb * 128 + tid - 128) * D_SZ);
  const int aoff = tid * 32;

  const unsigned int lds_w0 =
      (unsigned int)(uintptr_t)(void*)(smem + 32768);        // LDS byte addr of Wbuf0
  const unsigned int rem = (unsigned int)(C_SZ - cb * 64);   // >= 32
  const unsigned long long wg =
      (unsigned long long)(uintptr_t)(wH + (size_t)cb * 64 * D_SZ);

  v8f accf[2][4];
#pragma unroll
  for (int mf = 0; mf < 2; ++mf)
#pragma unroll
    for (int nf = 0; nf < 4; ++nf)
#pragma unroll
      for (int i = 0; i < 8; ++i) accf[mf][nf][i] = 0.f;

  // ---- prologue: fill buffer 0 ----
  {
    h8 r0 = *(const h8*)(asrc + 0);
    h8 r1 = *(const h8*)(asrc + 8);
    h8 r2 = *(const h8*)(asrc + 16);
    h8 r3 = *(const h8*)(asrc + 24);
    *(h8*)(Abase + aoff + 0)  = r0;
    *(h8*)(Abase + aoff + 8)  = r1;
    *(h8*)(Abase + aoff + 16) = r2;
    *(h8*)(Abase + aoff + 24) = r3;
    if (wv == 0) tdm_load_w_tile(lds_w0, wg, rem);
  }

#pragma unroll 1
  for (int ks = 0; ks < D_SZ / 32; ++ks) {
    const int cur = ks & 1, nxt = cur ^ 1;
    const bool have_next = (ks + 1 < D_SZ / 32);

    if (wv == 0) __builtin_amdgcn_s_wait_tensorcnt(0);  // W tile `cur` landed
    __syncthreads();  // publishes W(cur) + A(cur); prior computes done

    h8 r0, r1, r2, r3;
    if (have_next) {
      const int k1 = (ks + 1) * 32;
      r0 = *(const h8*)(asrc + k1 + 0);
      r1 = *(const h8*)(asrc + k1 + 8);
      r2 = *(const h8*)(asrc + k1 + 16);
      r3 = *(const h8*)(asrc + k1 + 24);
      if (wv == 0)
        tdm_load_w_tile(lds_w0 + (unsigned)nxt * 4096u,
                        wg + (unsigned long long)k1 * 2ull, rem);
      __builtin_prefetch(asrc + k1 + 32, 0, 0);  // global_prefetch_b8
    }

    const _Float16* Ac = Abase + cur * 8192;   // 16 KB / 2 B
    const _Float16* Wc = Wbase + cur * 2048;   //  4 KB / 2 B

    // A fragments (16x32 f16): lane row M=lr, K = lh*8+{0..7} and 16+lh*8+{0..7}
    v16h a[2];
#pragma unroll
    for (int mf = 0; mf < 2; ++mf) {
      const _Float16* base = Ac + (part * 128 + m0 + mf * 16 + lr) * 32 + lh * 8;
      union { v16h v; h8 h[2]; } u;
      u.h[0] = *(const h8*)(base);
      u.h[1] = *(const h8*)(base + 16);
      a[mf] = u.v;
    }
    // B fragments (32x16 f16): lane col N=lr, K = lh*8+{0..7} and 16+lh*8+{0..7}
    v16h bf[4];
#pragma unroll
    for (int nf = 0; nf < 4; ++nf) {
      const _Float16* bb = Wc + (nf * 16 + lr) * 32 + lh * 8;
      union { v16h v; h8 h[2]; } ub;
      ub.h[0] = *(const h8*)(bb);
      ub.h[1] = *(const h8*)(bb + 16);
      bf[nf] = ub.v;
    }

#pragma unroll
    for (int mf = 0; mf < 2; ++mf)
#pragma unroll
      for (int nf = 0; nf < 4; ++nf)
        accf[mf][nf] = __builtin_amdgcn_wmma_f32_16x16x32_f16(
            false, a[mf], false, bf[nf], (short)0, accf[mf][nf], false, false);

    if (have_next) {  // A(nxt) last read in step ks-1 -> safe after top barrier
      _Float16* An = Abase + nxt * 8192;
      *(h8*)(An + aoff + 0)  = r0;
      *(h8*)(An + aoff + 8)  = r1;
      *(h8*)(An + aoff + 16) = r2;
      *(h8*)(An + aoff + 24) = r3;
    }
  }

  __syncthreads();  // done with staged f16 tiles; reuse LDS as f32 result tile
  // D layout: lane L, vgpr r -> row = 8*(L/16)+r, col = L%16
#pragma unroll
  for (int mf = 0; mf < 2; ++mf)
#pragma unroll
    for (int nf = 0; nf < 4; ++nf) {
      const int rowb = part * 128 + m0 + mf * 16 + lh * 8;
#pragma unroll
      for (int r = 0; r < 8; ++r)
        Fsh[(rowb + r) * 64 + nf * 16 + lr] = accf[mf][nf][r];
    }
  __syncthreads();

  // Epilogue: 2 threads per batch row, 32 classes each.
  const int bl = tid >> 1, half = tid & 1;
  const int bg = rb * 128 + bl;
  const int lab = labels[bg];
  const float ctm = cosTM[bg];
  float s = 0.f;
#pragma unroll 1
  for (int j = 0; j < 32; ++j) {
    const int n = half * 32 + j;
    const int c = cb * 64 + n;
    if (c < C_SZ && c != lab) {
      const float lg  = fminf(fmaxf(Fsh[bl * 64 + n], -1.f), 1.f);
      const float sim = Fsh[(128 + bl) * 64 + n];
      float dn = sqrtf(fmaxf(2.f - 2.f * sim, 0.f));
      dn = fmaxf(dn, 1e-12f);
      s += __expf((lg - ctm) / dn);
    }
  }
  s += __shfl_xor(s, 1, 32);
  if (half == 0) atomicAdd(&acc[bg], s);
}

// ---------------------------------------------------------------------------
// Kernel 6: out = mean(log1p(acc)). Single block.
// ---------------------------------------------------------------------------
__global__ void finalize_kernel(const float* __restrict__ acc,
                                float* __restrict__ out) {
  __shared__ float red[256];
  const int t = threadIdx.x;
  red[t] = log1pf(acc[t]) + log1pf(acc[t + 256]);
  __syncthreads();
#pragma unroll
  for (int s2 = 128; s2 > 0; s2 >>= 1) {
    if (t < s2) red[t] += red[t + s2];
    __syncthreads();
  }
  if (t == 0) out[0] = red[0] / (float)B_SZ;
}

// ---------------------------------------------------------------------------
extern "C" void kernel_launch(void* const* d_in, const int* in_sizes, int n_in,
                              void* d_out, int out_size, void* d_ws, size_t ws_size,
                              hipStream_t stream) {
  const float* emb    = (const float*)d_in[0];   // (512, 512) f32
  const int*   labels = (const int*)d_in[1];     // (512,) i32
  const float* weight = (const float*)d_in[2];   // (100000, 512) f32
  float* out = (float*)d_out;

  char* ws = (char*)d_ws;
  // workspace layout (all 256B-aligned offsets)
  _Float16* wH   = (_Float16*)(ws);                         // 102,400,000 B
  float*    eN   = (float*)   (ws + 102400000);             //   1,048,576 B
  _Float16* eH   = (_Float16*)(ws + 103448576);             //     524,288 B
  _Float16* wmH  = (_Float16*)(ws + 103972864);             //     524,288 B
  float*    cosT = (float*)   (ws + 104497152);             //       2,048 B
  float*    acc  = (float*)   (ws + 104499200);             //       2,048 B

  prep_e_kernel <<<64,    256, 0, stream>>>(emb, eN, eH);
  prep_w_kernel <<<12500, 256, 0, stream>>>(weight, wH);
  prep_wm_kernel<<<64,    256, 0, stream>>>(eN, labels, wH, wmH, cosT);
  zero_acc_kernel<<<2,    256, 0, stream>>>(acc);

  dim3 grid(1563, 4);  // 1563*64 >= 100000 classes, 4 batch tiles of 128
  arc_gemm_kernel<<<grid, 256, 0, stream>>>(eH, wmH, wH, labels, cosT, acc);

  finalize_kernel<<<1, 256, 0, stream>>>(acc, out);
}
